// Attention_87531433492858
// MI455X (gfx1250) — matrix operand order repaired and checked by reading, not compiled, not run on hardware
//
#include <hip/hip_runtime.h>
#include <hip/hip_bf16.h>

// ---------------------------------------------------------------------------
// GPT-2 causal attention forward for gfx1250 (MI455X), bf16 WMMA + f32 acc.
//   B=4, S=2048, D=768, H=12, DH=64
// Pipeline: convert -> QKV gemm -> flash attention -> proj gemm.
// All matmuls use v_wmma_f32_16x16x32_bf16 (wave32).
// GEMM waves own 32x64 output tiles; attention waves own 32 query rows.
// ---------------------------------------------------------------------------

typedef __attribute__((ext_vector_type(16))) __bf16 v16bf;
typedef __attribute__((ext_vector_type(8)))  float  v8f;

#define WMMA_BF16(a, b, c) \
  __builtin_amdgcn_wmma_f32_16x16x32_bf16(false, (a), false, (b), (short)0, (c), false, false)

constexpr int Bb  = 4;
constexpr int Ss  = 2048;
constexpr int Dd  = 768;
constexpr int Hh  = 12;
constexpr int DHd = 64;
constexpr int D3  = 3 * Dd;   // 2304

// ---------------------------------------------------------------------------
// Fragment loaders (16x16x32 bf16 WMMA; contiguous k order chosen by this code, A and B match):
//  A (16x32, MxK): lanes 0-15 hold M=0..15 / K=0..15, lanes 16-31 K=16..31;
//    VGPR v holds K pair {v*2, v*2+1}, +16 per lane-half.
//  B (32x16, KxN): lane n = N, lanes 0-15 hold K=0..15, lanes 16-31 K=16..31,
//    consecutive elements = consecutive K (mirrors the SWMMAC B description).
//  C/D (16x16 f32): lane n = N (mod 16); VGPR r holds row M = r + 8*(lane/16).
// ---------------------------------------------------------------------------

__device__ __forceinline__ v16bf load_frag_a(const __bf16* __restrict__ src,
                                             int ld, int lane) {
  // src points at element (m=0, k=0) of a 16x32 tile, row-major, stride ld.
  const int m  = lane & 15;
  const int hf = lane >> 4;
  const __bf16* row = src + (size_t)m * ld + hf * 16;
  v16bf a;
#pragma unroll
  for (int v = 0; v < 8; ++v) {
    const int k = v * 2;
    a[2 * v]     = row[k];
    a[2 * v + 1] = row[k + 1];
  }
  return a;
}

__device__ __forceinline__ v16bf load_frag_b(const __bf16* __restrict__ src,
                                             int ld, int lane) {
  // src points at element (n=0, k=0); storage is N-major: src[n*ld + k].
  const int n  = lane & 15;
  const int hf = lane >> 4;
  const __bf16* col = src + (size_t)n * ld + hf * 16;
  v16bf b;
#pragma unroll
  for (int i = 0; i < 16; ++i) b[i] = col[i];
  return b;
}

__device__ __forceinline__ float rowmax16(float v) {
#pragma unroll
  for (int m = 8; m >= 1; m >>= 1) v = fmaxf(v, __shfl_xor(v, m, 32));
  return v;
}
__device__ __forceinline__ float rowsum16(float v) {
#pragma unroll
  for (int m = 8; m >= 1; m >>= 1) v += __shfl_xor(v, m, 32);
  return v;
}

// ---------------------------------------------------------------------------
// Conversion / transpose kernels
// ---------------------------------------------------------------------------
__global__ void f32_to_bf16_kernel(const float* __restrict__ src,
                                   __bf16* __restrict__ dst, long n) {
  long i = (long)blockIdx.x * blockDim.x + threadIdx.x;
  long stride = (long)gridDim.x * blockDim.x;
  for (; i < n; i += stride) dst[i] = (__bf16)src[i];
}

// dst[n*rows + k] = src[k*cols + n]  (rows = K dim, cols = N dim)
__global__ void transpose_bf16_kernel(const float* __restrict__ src,
                                      __bf16* __restrict__ dst,
                                      int rows, int cols) {
  long i = (long)blockIdx.x * blockDim.x + threadIdx.x;
  if (i >= (long)rows * cols) return;
  int n = (int)(i / rows);
  int k = (int)(i % rows);
  dst[i] = (__bf16)src[(size_t)k * cols + n];
}

// ---------------------------------------------------------------------------
// QKV GEMM: qkv[m, n] = x[m, :] . W[:, n] + b[n];  m in [0, B*S), n in [0, 3D)
// Wave tile 32x64 (2 A-frags reuse 4 B-frags -> 8 WMMA / k-step),
// block = 4 waves = 32x256. Epilogue adds bias and scatters bf16 into
// Q[b,h,s,dh], K[b,h,s,dh], V^T[b,h,dh,s].
// ---------------------------------------------------------------------------
__global__ __launch_bounds__(128) void qkv_gemm_kernel(
    const __bf16* __restrict__ xbf, const __bf16* __restrict__ wt,
    const float* __restrict__ bias,
    __bf16* __restrict__ qb, __bf16* __restrict__ kb, __bf16* __restrict__ vt) {
  const int lane  = threadIdx.x & 31;
  const int wv    = threadIdx.x >> 5;
  const int mBase = blockIdx.y * 32;
  const int nBase = blockIdx.x * 256 + wv * 64;

  v8f acc[2][4] = {};
  for (int kc = 0; kc < Dd; kc += 32) {
    const v16bf a0 = load_frag_a(xbf + (size_t)mBase * Dd + kc, Dd, lane);
    const v16bf a1 = load_frag_a(xbf + (size_t)(mBase + 16) * Dd + kc, Dd, lane);
#pragma unroll
    for (int nt = 0; nt < 4; ++nt) {
      v16bf b = load_frag_b(wt + (size_t)(nBase + nt * 16) * Dd + kc, Dd, lane);
      acc[0][nt] = WMMA_BF16(a0, b, acc[0][nt]);
      acc[1][nt] = WMMA_BF16(a1, b, acc[1][nt]);
    }
  }

  const int ln = lane & 15, hf = lane >> 4;
#pragma unroll
  for (int nt = 0; nt < 4; ++nt) {
    const int   n     = nBase + nt * 16 + ln;
    const float bv    = bias[n];
    const int   which = n / Dd;
    const int   rem   = n % Dd;
    const int   h     = rem / DHd;
    const int   dh    = rem % DHd;
#pragma unroll
    for (int mh = 0; mh < 2; ++mh) {
#pragma unroll
      for (int r = 0; r < 8; ++r) {
        const int m  = mBase + mh * 16 + r + 8 * hf;
        const int bi = m / Ss;
        const int s  = m % Ss;
        const __bf16 val = (__bf16)(acc[mh][nt][r] + bv);
        if (which == 0)
          qb[((size_t)(bi * Hh + h) * Ss + s) * DHd + dh] = val;
        else if (which == 1)
          kb[((size_t)(bi * Hh + h) * Ss + s) * DHd + dh] = val;
        else
          vt[((size_t)(bi * Hh + h) * DHd + dh) * Ss + s] = val;
      }
    }
  }
}

// ---------------------------------------------------------------------------
// Flash attention: one wave owns one (b, h, 32-query tile); streams causal
// 32-key chunks with online softmax. K/V B-fragments are reused across both
// 16-row halves of the Q tile (16 WMMA per 8 fragment loads per chunk).
// No 1/sqrt(dh) scale (reference applies none).
// ---------------------------------------------------------------------------
__global__ __launch_bounds__(128) void attn_kernel(
    const __bf16* __restrict__ qb, const __bf16* __restrict__ kb,
    const __bf16* __restrict__ vt, __bf16* __restrict__ ob) {
  __shared__ __bf16 plds[4][32 * 32];  // per-wave P staging (C-layout -> A-layout)

  const int lane = threadIdx.x & 31;
  const int wv   = threadIdx.x >> 5;
  const int w    = blockIdx.x * 4 + wv;

  const int QT = Ss / 32;               // 64 query tiles per (b, h)
  const int qt = w % QT;
  const int hh = (w / QT) % Hh;
  const int bb = w / (QT * Hh);
  const int q0 = qt * 32;

  const __bf16* Qp = qb + ((size_t)(bb * Hh + hh) * Ss + q0) * DHd;
  const __bf16* Kp = kb + (size_t)(bb * Hh + hh) * Ss * DHd;
  const __bf16* Vp = vt + (size_t)(bb * Hh + hh) * DHd * Ss;

  v16bf aq[2][2];  // [m-half][k-chunk]
#pragma unroll
  for (int mh = 0; mh < 2; ++mh)
#pragma unroll
    for (int kc = 0; kc < 2; ++kc)
      aq[mh][kc] = load_frag_a(Qp + (size_t)(mh * 16) * DHd + kc * 32, DHd, lane);

  float mrow[2][8], lrow[2][8];
#pragma unroll
  for (int mh = 0; mh < 2; ++mh)
#pragma unroll
    for (int r = 0; r < 8; ++r) { mrow[mh][r] = -1.0e30f; lrow[mh][r] = 0.0f; }
  v8f acc[2][4] = {};

  const int ln = lane & 15, hf = lane >> 4;
  __bf16* pw = &plds[wv][0];

  const int kvEnd = q0 + 32;  // causal: keys <= last query row of this tile
  for (int kv0 = 0; kv0 < kvEnd; kv0 += 32) {
    if (kv0 + 32 < kvEnd)  // prefetch next K chunk (global_prefetch_b8)
      __builtin_prefetch(Kp + (size_t)(kv0 + 32) * DHd, 0, 1);

    // ---- scores S = Q K^T: 32 rows x 32 key columns (8 WMMAs, 4 K-frags) ---
    v16bf bk[2][2];  // [n-tile][k-chunk]
#pragma unroll
    for (int nt = 0; nt < 2; ++nt)
#pragma unroll
      for (int kc = 0; kc < 2; ++kc)
        bk[nt][kc] =
            load_frag_b(Kp + (size_t)(kv0 + nt * 16) * DHd + kc * 32, DHd, lane);

    v8f sc[2][2];  // [m-half][n-tile]
#pragma unroll
    for (int mh = 0; mh < 2; ++mh) {
#pragma unroll
      for (int nt = 0; nt < 2; ++nt) {
        v8f s = {};
        s = WMMA_BF16(aq[mh][0], bk[nt][0], s);
        s = WMMA_BF16(aq[mh][1], bk[nt][1], s);
        sc[mh][nt] = s;
      }
    }

    // ---- causal mask + online softmax (row lives in one 16-lane group) ----
    const int k0p = kv0 + ln;
    const int k1p = kv0 + 16 + ln;
#pragma unroll
    for (int mh = 0; mh < 2; ++mh) {
#pragma unroll
      for (int r = 0; r < 8; ++r) {
        const int   qpos = q0 + mh * 16 + r + 8 * hf;
        const float a0   = (k0p <= qpos) ? sc[mh][0][r] : -1.0e30f;
        const float a1   = (k1p <= qpos) ? sc[mh][1][r] : -1.0e30f;
        const float cmax = rowmax16(fmaxf(a0, a1));
        const float mnew = fmaxf(mrow[mh][r], cmax);
        const float alpha = __expf(mrow[mh][r] - mnew);
        const float p0 = __expf(a0 - mnew);
        const float p1 = __expf(a1 - mnew);
        lrow[mh][r] = lrow[mh][r] * alpha + rowsum16(p0 + p1);
        mrow[mh][r] = mnew;
#pragma unroll
        for (int dt = 0; dt < 4; ++dt) acc[mh][dt][r] = acc[mh][dt][r] * alpha;
        // stage P (bf16, row-major 32x32) for A-fragment reload
        const int base = (mh * 16 + r + 8 * hf) * 32;
        pw[base + ln]      = (__bf16)p0;
        pw[base + 16 + ln] = (__bf16)p1;
      }
    }

    // ---- O += P V  (A from LDS, B from V^T; 8 WMMAs, 4 V-frags) ----
    v16bf bv[4];
#pragma unroll
    for (int dt = 0; dt < 4; ++dt)
      bv[dt] = load_frag_b(Vp + (size_t)(dt * 16) * Ss + kv0, Ss, lane);
#pragma unroll
    for (int mh = 0; mh < 2; ++mh) {
      const v16bf pa = load_frag_a(pw + mh * 16 * 32, 32, lane);
#pragma unroll
      for (int dt = 0; dt < 4; ++dt) acc[mh][dt] = WMMA_BF16(pa, bv[dt], acc[mh][dt]);
    }
  }

  // ---- normalize and store merged-head bf16 output [B, S, D] ----
#pragma unroll
  for (int mh = 0; mh < 2; ++mh) {
    float linv[8];
#pragma unroll
    for (int r = 0; r < 8; ++r) linv[r] = 1.0f / lrow[mh][r];
#pragma unroll
    for (int dt = 0; dt < 4; ++dt) {
      const int dh = dt * 16 + ln;
#pragma unroll
      for (int r = 0; r < 8; ++r) {
        const int s = q0 + mh * 16 + r + 8 * hf;
        ob[(size_t)(bb * Ss + s) * Dd + hh * DHd + dh] =
            (__bf16)(acc[mh][dt][r] * linv[r]);
      }
    }
  }
}

// ---------------------------------------------------------------------------
// Output projection: out[m, n] = attn[m, :] . Wp[:, n] + bp[n], f32 output.
// Wave tile 32x64.
// ---------------------------------------------------------------------------
__global__ __launch_bounds__(128) void proj_gemm_kernel(
    const __bf16* __restrict__ ob, const __bf16* __restrict__ wpt,
    const float* __restrict__ bias, float* __restrict__ out) {
  const int lane  = threadIdx.x & 31;
  const int wv    = threadIdx.x >> 5;
  const int mBase = blockIdx.y * 32;
  const int nBase = blockIdx.x * 256 + wv * 64;

  v8f acc[2][4] = {};
  for (int kc = 0; kc < Dd; kc += 32) {
    const v16bf a0 = load_frag_a(ob + (size_t)mBase * Dd + kc, Dd, lane);
    const v16bf a1 = load_frag_a(ob + (size_t)(mBase + 16) * Dd + kc, Dd, lane);
#pragma unroll
    for (int nt = 0; nt < 4; ++nt) {
      v16bf b = load_frag_b(wpt + (size_t)(nBase + nt * 16) * Dd + kc, Dd, lane);
      acc[0][nt] = WMMA_BF16(a0, b, acc[0][nt]);
      acc[1][nt] = WMMA_BF16(a1, b, acc[1][nt]);
    }
  }

  const int ln = lane & 15, hf = lane >> 4;
#pragma unroll
  for (int nt = 0; nt < 4; ++nt) {
    const int   n  = nBase + nt * 16 + ln;
    const float bv = bias[n];
#pragma unroll
    for (int mh = 0; mh < 2; ++mh) {
#pragma unroll
      for (int r = 0; r < 8; ++r) {
        const int m = mBase + mh * 16 + r + 8 * hf;
        out[(size_t)m * Dd + n] = acc[mh][nt][r] + bv;
      }
    }
  }
}

// ---------------------------------------------------------------------------
// Launch
// ---------------------------------------------------------------------------
extern "C" void kernel_launch(void* const* d_in, const int* in_sizes, int n_in,
                              void* d_out, int out_size, void* d_ws,
                              size_t ws_size, hipStream_t stream) {
  const float* x      = (const float*)d_in[0];
  const float* W_attn = (const float*)d_in[1];
  const float* b_attn = (const float*)d_in[2];
  const float* W_proj = (const float*)d_in[3];
  const float* b_proj = (const float*)d_in[4];
  float*       out    = (float*)d_out;

  const size_t NX = (size_t)Bb * Ss * Dd;  // 6,291,456
  const size_t NW = (size_t)D3 * Dd;       // 1,769,472
  const size_t NP = (size_t)Dd * Dd;       //   589,824

  char* p = (char*)d_ws;
  __bf16* xbf = (__bf16*)p; p += NX * 2;   // x in bf16
  __bf16* wat = (__bf16*)p; p += NW * 2;   // W_attn^T (N-major) bf16
  __bf16* wpt = (__bf16*)p; p += NP * 2;   // W_proj^T (N-major) bf16
  __bf16* qb  = (__bf16*)p; p += NX * 2;   // Q  [B,H,S,DH]
  __bf16* kb  = (__bf16*)p; p += NX * 2;   // K  [B,H,S,DH]
  __bf16* vt  = (__bf16*)p; p += NX * 2;   // V^T[B,H,DH,S]
  __bf16* ob  = (__bf16*)p; p += NX * 2;   // attn out [B,S,D]
  (void)ws_size; (void)in_sizes; (void)n_in; (void)out_size;

  f32_to_bf16_kernel<<<4096, 256, 0, stream>>>(x, xbf, (long)NX);
  transpose_bf16_kernel<<<(int)((NW + 255) / 256), 256, 0, stream>>>(
      W_attn, wat, Dd, D3);
  transpose_bf16_kernel<<<(int)((NP + 255) / 256), 256, 0, stream>>>(
      W_proj, wpt, Dd, Dd);

  qkv_gemm_kernel<<<dim3(D3 / 256, (Bb * Ss) / 32), 128, 0, stream>>>(
      xbf, wat, b_attn, qb, kb, vt);

  attn_kernel<<<(Bb * Hh * (Ss / 32)) / 4, 128, 0, stream>>>(qb, kb, vt, ob);

  proj_gemm_kernel<<<dim3(Dd / 256, (Bb * Ss) / 32), 128, 0, stream>>>(
      ob, wpt, b_proj, out);
}
